// NonLocalWithCoefNet_41042707480841
// MI455X (gfx1250) — matrix operand order repaired and checked by reading, not compiled, not run on hardware
//
#include <hip/hip_runtime.h>
#include <hip/hip_bf16.h>
#include <math.h>

// ---------------- problem constants ----------------
#define NB   4
#define CCH  256
#define ICH  128
#define HWP  4096   // 64*64
#define SCALE_QK 0.08838834764831845f  // 1/sqrt(128)

typedef _Float16 f16;
typedef __attribute__((ext_vector_type(16))) _Float16 v16h;
typedef __attribute__((ext_vector_type(8)))  _Float16 v8h;
typedef __attribute__((ext_vector_type(8)))  float    v8f;

// ---------------- WMMA helpers ----------------
__device__ __forceinline__ v8f wmma_f16(v16h a, v16h b, v8f c) {
  // D = A(16x32 f16) * B(32x16 f16) + C(16x16 f32)
  return __builtin_amdgcn_wmma_f32_16x16x32_f16(
      /*neg_a=*/false, a, /*neg_b=*/false, b,
      /*c_mod=*/(short)0, c, /*reuse_a=*/false, /*reuse_b=*/false);
}

// A fragment 16x32 f16 from row-major [row][ld] storage.
// lane L: row = base_row + L%16; K(e) = (e/8)*16 + (L/16)*8 + e%8
__device__ __forceinline__ v16h load_frag_a16(const f16* __restrict__ base, int row,
                                              int ld, int kk, int lane) {
  const int r   = row + (lane & 15);
  const int sel = (lane >> 4) * 8;
  const f16* rp = base + (size_t)r * ld + kk + sel;
  v8h lo = *(const v8h*)(rp);        // K block [kk+sel .. +7]
  v8h hi = *(const v8h*)(rp + 16);   // K block [kk+16+sel .. +7]
  return __builtin_shufflevector(lo, hi, 0,1,2,3,4,5,6,7,8,9,10,11,12,13,14,15);
}

// B fragment 32x16 f16 from column-major-for-B storage [col][ld] (K contiguous).
// lane L: col = base_col + L%16; K(e) = (e/8)*16 + (L/16)*8 + e%8
__device__ __forceinline__ v16h load_frag_b16(const f16* __restrict__ base, int col,
                                              int ld, int kk, int lane) {
  const int c   = col + (lane & 15);
  const int sel = (lane >> 4) * 8;
  const f16* cp = base + (size_t)c * ld + kk + sel;
  v8h lo = *(const v8h*)(cp);
  v8h hi = *(const v8h*)(cp + 16);
  return __builtin_shufflevector(lo, hi, 0,1,2,3,4,5,6,7,8,9,10,11,12,13,14,15);
}

// A fragment from f32 row-major [row][ld] storage, converted to f16 in-register.
__device__ __forceinline__ v16h load_frag_a_f32(const float* __restrict__ base, int row,
                                                int ld, int kk, int lane) {
  const int r   = row + (lane & 15);
  const int sel = (lane >> 4) * 8;
  const float* rp = base + (size_t)r * ld + kk + sel;
  float4 a0 = *(const float4*)(rp);
  float4 a1 = *(const float4*)(rp + 4);
  float4 b0 = *(const float4*)(rp + 16);
  float4 b1 = *(const float4*)(rp + 20);
  v16h v;
  v[0]=(f16)a0.x;  v[1]=(f16)a0.y;  v[2]=(f16)a0.z;  v[3]=(f16)a0.w;
  v[4]=(f16)a1.x;  v[5]=(f16)a1.y;  v[6]=(f16)a1.z;  v[7]=(f16)a1.w;
  v[8]=(f16)b0.x;  v[9]=(f16)b0.y;  v[10]=(f16)b0.z; v[11]=(f16)b0.w;
  v[12]=(f16)b1.x; v[13]=(f16)b1.y; v[14]=(f16)b1.z; v[15]=(f16)b1.w;
  return v;
}

// ---------------- 1x1 projections ----------------
// out[n][p][ICH] = f16( W[cout][C] . x[n][:][p] + b[cout] )   (theta / phi)
__global__ __launch_bounds__(256) void proj_pc_kernel(const float* __restrict__ x,
                                                      const float* __restrict__ w,
                                                      const float* __restrict__ b,
                                                      f16* __restrict__ out) {
  const int n = blockIdx.z;
  const int c = threadIdx.x & 127;
  const int p = blockIdx.x * 2 + (threadIdx.x >> 7);
  const float* xn = x + (size_t)n * CCH * HWP;
  const float* wr = w + (size_t)c * CCH;
  float acc = b[c];
  #pragma unroll 8
  for (int ci = 0; ci < CCH; ++ci)
    acc += wr[ci] * xn[(size_t)ci * HWP + p];
  out[(size_t)n * HWP * ICH + (size_t)p * ICH + c] = (f16)acc;
}

// out[n][ICH][p] = f16( W[cout][C] . x[n][:][p] + b[cout] )   (g)
__global__ __launch_bounds__(256) void proj_cp_kernel(const float* __restrict__ x,
                                                      const float* __restrict__ w,
                                                      const float* __restrict__ b,
                                                      f16* __restrict__ out) {
  const int n = blockIdx.z;
  const int c = blockIdx.y;
  const int p = blockIdx.x * 256 + threadIdx.x;
  const float* xn = x + (size_t)n * CCH * HWP;
  const float* wr = w + (size_t)c * CCH;
  float acc = b[c];
  #pragma unroll 8
  for (int ci = 0; ci < CCH; ++ci)
    acc += wr[ci] * xn[(size_t)ci * HWP + p];
  out[(size_t)n * ICH * HWP + (size_t)c * HWP + p] = (f16)acc;
}

// ---------------- scores = (theta^T phi) / sqrt(128), WMMA ----------------
// theta: [n][q][128] f16, phi: [n][k][128] f16, S: [n][4096][4096] f32
// WG = 256 thr = 8 waves, tile 128(M) x 128(N); wave -> 32x64; K loop = 128/32
__global__ __launch_bounds__(256) void scores_kernel(const f16* __restrict__ theta,
                                                     const f16* __restrict__ phi,
                                                     float* __restrict__ S) {
  const int n    = blockIdx.z;
  const int lane = threadIdx.x & 31;
  const int wave = threadIdx.x >> 5;
  const int q0   = blockIdx.y * 128 + (wave >> 1) * 32;
  const int k0   = blockIdx.x * 128 + (wave & 1) * 64;
  const f16* A = theta + (size_t)n * HWP * ICH;
  const f16* B = phi   + (size_t)n * HWP * ICH;

  v8f zero = {};
  v8f acc[2][4];
  #pragma unroll
  for (int i = 0; i < 2; ++i)
    #pragma unroll
    for (int j = 0; j < 4; ++j) acc[i][j] = zero;

  #pragma unroll
  for (int kk = 0; kk < ICH; kk += 32) {
    v16h a0 = load_frag_a16(A, q0,      ICH, kk, lane);
    v16h a1 = load_frag_a16(A, q0 + 16, ICH, kk, lane);
    v16h b0 = load_frag_b16(B, k0 +  0, ICH, kk, lane);
    v16h b1 = load_frag_b16(B, k0 + 16, ICH, kk, lane);
    v16h b2 = load_frag_b16(B, k0 + 32, ICH, kk, lane);
    v16h b3 = load_frag_b16(B, k0 + 48, ICH, kk, lane);
    acc[0][0] = wmma_f16(a0, b0, acc[0][0]);
    acc[0][1] = wmma_f16(a0, b1, acc[0][1]);
    acc[0][2] = wmma_f16(a0, b2, acc[0][2]);
    acc[0][3] = wmma_f16(a0, b3, acc[0][3]);
    acc[1][0] = wmma_f16(a1, b0, acc[1][0]);
    acc[1][1] = wmma_f16(a1, b1, acc[1][1]);
    acc[1][2] = wmma_f16(a1, b2, acc[1][2]);
    acc[1][3] = wmma_f16(a1, b3, acc[1][3]);
  }

  float* Sn = S + (size_t)n * HWP * HWP;
  const int rbase = (lane >> 4) * 8;
  const int cbase = (lane & 15);
  #pragma unroll
  for (int i = 0; i < 2; ++i)
    #pragma unroll
    for (int j = 0; j < 4; ++j) {
      const int col = k0 + j * 16 + cbase;
      #pragma unroll
      for (int v = 0; v < 8; ++v)
        Sn[(size_t)(q0 + i * 16 + rbase + v) * HWP + col] = acc[i][j][v] * SCALE_QK;
    }
}

// ---------------- row softmax, in place on [rows=NB*HWP][4096] ----------------
__global__ __launch_bounds__(256) void softmax_kernel(float* __restrict__ S) {
  float* row = S + (size_t)blockIdx.x * HWP;
  const int t = threadIdx.x;
  __shared__ float red[8];
  __shared__ float bval;

  float v[16];
  float m = -3.402823466e38f;
  #pragma unroll
  for (int i = 0; i < 16; ++i) { v[i] = row[t + i * 256]; m = fmaxf(m, v[i]); }
  #pragma unroll
  for (int o = 16; o > 0; o >>= 1) m = fmaxf(m, __shfl_xor(m, o, 32));
  if ((t & 31) == 0) red[t >> 5] = m;
  __syncthreads();
  if (t == 0) {
    float mm = red[0];
    #pragma unroll
    for (int i = 1; i < 8; ++i) mm = fmaxf(mm, red[i]);
    bval = mm;
  }
  __syncthreads();
  m = bval;

  float s = 0.f;
  #pragma unroll
  for (int i = 0; i < 16; ++i) { v[i] = __expf(v[i] - m); s += v[i]; }
  #pragma unroll
  for (int o = 16; o > 0; o >>= 1) s += __shfl_xor(s, o, 32);
  __syncthreads();
  if ((t & 31) == 0) red[t >> 5] = s;
  __syncthreads();
  if (t == 0) {
    float ss = 0.f;
    #pragma unroll
    for (int i = 0; i < 8; ++i) ss += red[i];
    bval = ss;
  }
  __syncthreads();
  const float inv = 1.0f / bval;
  #pragma unroll
  for (int i = 0; i < 16; ++i) row[t + i * 256] = v[i] * inv;
}

// ---------------- y = P * g, WMMA (P f32 -> f16 in-register) ----------------
// P: [n][4096][4096] f32 row-major; g: [n][128][4096] f16 (K=k contiguous)
// Y: [n][q][128] f32. WG tile 128(M) x 128(N=full ICH), K loop over 4096.
__global__ __launch_bounds__(256) void ygemm_kernel(const float* __restrict__ P,
                                                    const f16* __restrict__ g,
                                                    float* __restrict__ Y) {
  const int n    = blockIdx.z;
  const int lane = threadIdx.x & 31;
  const int wave = threadIdx.x >> 5;
  const int q0   = blockIdx.y * 128 + (wave >> 1) * 32;
  const int c0   = (wave & 1) * 64;
  const float* Pn = P + (size_t)n * HWP * HWP;
  const f16*   G  = g + (size_t)n * ICH * HWP;

  v8f zero = {};
  v8f acc[2][4];
  #pragma unroll
  for (int i = 0; i < 2; ++i)
    #pragma unroll
    for (int j = 0; j < 4; ++j) acc[i][j] = zero;

  #pragma unroll 2
  for (int kk = 0; kk < HWP; kk += 32) {
    // prefetch next K-slab of P (L2-resident g needs none)
    if (kk + 128 < HWP)
      __builtin_prefetch(Pn + (size_t)(q0 + (lane & 15)) * HWP + kk + 128, 0, 0);
    v16h a0 = load_frag_a_f32(Pn, q0,      HWP, kk, lane);
    v16h a1 = load_frag_a_f32(Pn, q0 + 16, HWP, kk, lane);
    v16h b0 = load_frag_b16(G, c0 +  0, HWP, kk, lane);
    v16h b1 = load_frag_b16(G, c0 + 16, HWP, kk, lane);
    v16h b2 = load_frag_b16(G, c0 + 32, HWP, kk, lane);
    v16h b3 = load_frag_b16(G, c0 + 48, HWP, kk, lane);
    acc[0][0] = wmma_f16(a0, b0, acc[0][0]);
    acc[0][1] = wmma_f16(a0, b1, acc[0][1]);
    acc[0][2] = wmma_f16(a0, b2, acc[0][2]);
    acc[0][3] = wmma_f16(a0, b3, acc[0][3]);
    acc[1][0] = wmma_f16(a1, b0, acc[1][0]);
    acc[1][1] = wmma_f16(a1, b1, acc[1][1]);
    acc[1][2] = wmma_f16(a1, b2, acc[1][2]);
    acc[1][3] = wmma_f16(a1, b3, acc[1][3]);
  }

  float* Yn = Y + (size_t)n * HWP * ICH;
  const int rbase = (lane >> 4) * 8;
  const int cbase = (lane & 15);
  #pragma unroll
  for (int i = 0; i < 2; ++i)
    #pragma unroll
    for (int j = 0; j < 4; ++j) {
      const int col = c0 + j * 16 + cbase;
      #pragma unroll
      for (int v = 0; v < 8; ++v)
        Yn[(size_t)(q0 + i * 16 + rbase + v) * ICH + col] = acc[i][j][v];
    }
}

// ---------------- out conv 1x1: 128 -> 256 ----------------
// yq: [n][p][128] f32 ; yc: [n][256][p] f32
__global__ __launch_bounds__(256) void outconv_kernel(const float* __restrict__ yq,
                                                      const float* __restrict__ w,
                                                      const float* __restrict__ b,
                                                      float* __restrict__ yc) {
  const int n = blockIdx.z;
  const int c = blockIdx.y;
  const int p = blockIdx.x * 256 + threadIdx.x;
  const float* yr = yq + (size_t)n * HWP * ICH + (size_t)p * ICH;
  const float* wr = w + (size_t)c * ICH;
  float acc = b[c];
  #pragma unroll 8
  for (int ic = 0; ic < ICH; ++ic) acc += wr[ic] * yr[ic];
  yc[(size_t)n * CCH * HWP + (size_t)c * HWP + p] = acc;
}

// ---------------- c1: 1x1 over concat([x, yc]) -> 256, relu ----------------
__global__ __launch_bounds__(256) void c1_kernel(const float* __restrict__ x,
                                                 const float* __restrict__ yc,
                                                 const float* __restrict__ w,
                                                 const float* __restrict__ b,
                                                 float* __restrict__ out) {
  const int n  = blockIdx.z;
  const int co = blockIdx.y;
  const int p  = blockIdx.x * 256 + threadIdx.x;
  const float* xn = x  + (size_t)n * CCH * HWP;
  const float* yn = yc + (size_t)n * CCH * HWP;
  const float* wr = w + (size_t)co * (2 * CCH);
  float acc = b[co];
  #pragma unroll 8
  for (int ci = 0; ci < CCH; ++ci) acc += wr[ci] * xn[(size_t)ci * HWP + p];
  #pragma unroll 8
  for (int ci = 0; ci < CCH; ++ci) acc += wr[CCH + ci] * yn[(size_t)ci * HWP + p];
  out[(size_t)n * CCH * HWP + (size_t)co * HWP + p] = fmaxf(acc, 0.f);
}

// ---------------- generic 3x3 conv, pad 1, optional relu ----------------
// in: [n][cin][64][64]; out: [n][gridDim.y][64][64]
__global__ __launch_bounds__(256) void conv3_kernel(const float* __restrict__ in,
                                                    const float* __restrict__ w,
                                                    const float* __restrict__ b,
                                                    float* __restrict__ out,
                                                    int cin, int do_relu) {
  const int n  = blockIdx.z;
  const int co = blockIdx.y;
  const int p  = blockIdx.x * 256 + threadIdx.x;
  const int h  = p >> 6;
  const int wx = p & 63;
  const float* inn = in + (size_t)n * cin * HWP;
  float acc = b[co];
  for (int ci = 0; ci < cin; ++ci) {
    const float* base = inn + (size_t)ci * HWP;
    const float* wk = w + ((size_t)co * cin + ci) * 9;
    #pragma unroll
    for (int ky = -1; ky <= 1; ++ky) {
      const int hh = h + ky;
      if (hh < 0 || hh >= 64) continue;
      #pragma unroll
      for (int kx = -1; kx <= 1; ++kx) {
        const int ww = wx + kx;
        if (ww < 0 || ww >= 64) continue;
        acc += wk[(ky + 1) * 3 + (kx + 1)] * base[hh * 64 + ww];
      }
    }
  }
  if (do_relu) acc = fmaxf(acc, 0.f);
  out[(size_t)n * gridDim.y * HWP + (size_t)co * HWP + p] = acc;
}

// ---------------- final: out = x + coef * yc ----------------
__global__ __launch_bounds__(256) void final_kernel(const float* __restrict__ x,
                                                    const float* __restrict__ yc,
                                                    const float* __restrict__ coef,
                                                    float* __restrict__ out) {
  const int n = blockIdx.z;
  const int c = blockIdx.y;
  const int p = blockIdx.x * 256 + threadIdx.x;
  const size_t idx = (size_t)n * CCH * HWP + (size_t)c * HWP + p;
  out[idx] = x[idx] + coef[(size_t)n * HWP + p] * yc[idx];
}

// ---------------- launch ----------------
extern "C" void kernel_launch(void* const* d_in, const int* in_sizes, int n_in,
                              void* d_out, int out_size, void* d_ws, size_t ws_size,
                              hipStream_t stream) {
  const float* x       = (const float*)d_in[0];
  const float* x_ref   = (const float*)d_in[1];
  const float* g_w     = (const float*)d_in[2];
  const float* g_b     = (const float*)d_in[3];
  const float* theta_w = (const float*)d_in[4];
  const float* theta_b = (const float*)d_in[5];
  const float* phi_w   = (const float*)d_in[6];
  const float* phi_b   = (const float*)d_in[7];
  const float* out_w   = (const float*)d_in[8];
  const float* out_b   = (const float*)d_in[9];
  const float* c1_w    = (const float*)d_in[10];
  const float* c1_b    = (const float*)d_in[11];
  const float* c2_w    = (const float*)d_in[12];
  const float* c2_b    = (const float*)d_in[13];
  const float* c3_w    = (const float*)d_in[14];
  const float* c3_b    = (const float*)d_in[15];
  const float* c4_w    = (const float*)d_in[16];
  const float* c4_b    = (const float*)d_in[17];

  float* outp = (float*)d_out;                             // [4,256,64,64]
  float* pair = outp + (size_t)NB * CCH * HWP;             // [4,4096,4096]

  // workspace layout (all regions 256B-aligned by construction)
  char* ws = (char*)d_ws;
  const size_t SZ_PROJ16 = (size_t)NB * HWP * ICH * sizeof(f16);   // 4 MB
  f16*   theta_h = (f16*)(ws);
  f16*   phi_h   = (f16*)(ws + SZ_PROJ16);
  f16*   g_h     = (f16*)(ws + 2 * SZ_PROJ16);
  float* yq      = (float*)(ws + 3 * SZ_PROJ16);                            // [n][q][128]
  float* yc      = (float*)((char*)yq + (size_t)NB * HWP * ICH * 4);        // [n][256][p]
  float* coef1   = (float*)((char*)yc + (size_t)NB * CCH * HWP * 4);        // [n][256][p]
  float* coef2   = (float*)((char*)coef1 + (size_t)NB * CCH * HWP * 4);     // [n][16][p]
  float* coef3   = (float*)((char*)coef2 + (size_t)NB * 16 * HWP * 4);      // [n][3][p]
  float* coef4   = (float*)((char*)coef3 + (size_t)NB * 3 * HWP * 4);       // [n][p]

  // 1) projections
  proj_pc_kernel<<<dim3(HWP / 2, 1, NB), 256, 0, stream>>>(x,     theta_w, theta_b, theta_h);
  proj_pc_kernel<<<dim3(HWP / 2, 1, NB), 256, 0, stream>>>(x_ref, phi_w,   phi_b,   phi_h);
  proj_cp_kernel<<<dim3(HWP / 256, ICH, NB), 256, 0, stream>>>(x_ref, g_w, g_b, g_h);

  // 2) scores (WMMA) -> raw scaled scores straight into pairwise output region
  scores_kernel<<<dim3(HWP / 128, HWP / 128, NB), 256, 0, stream>>>(theta_h, phi_h, pair);

  // 3) row softmax in place
  softmax_kernel<<<dim3(NB * HWP, 1, 1), 256, 0, stream>>>(pair);

  // 4) y = P * g (WMMA, P converted f32->f16 in-register)
  ygemm_kernel<<<dim3(1, HWP / 128, NB), 256, 0, stream>>>(pair, g_h, yq);

  // 5) out conv 1x1 (128 -> 256)
  outconv_kernel<<<dim3(HWP / 256, CCH, NB), 256, 0, stream>>>(yq, out_w, out_b, yc);

  // 6) c1 (1x1 over concat) + relu
  c1_kernel<<<dim3(HWP / 256, CCH, NB), 256, 0, stream>>>(x, yc, c1_w, c1_b, coef1);

  // 7) c2/c3/c4 3x3 convs
  conv3_kernel<<<dim3(HWP / 256, 16, NB), 256, 0, stream>>>(coef1, c2_w, c2_b, coef2, 256, 1);
  conv3_kernel<<<dim3(HWP / 256, 3,  NB), 256, 0, stream>>>(coef2, c3_w, c3_b, coef3, 16,  1);
  conv3_kernel<<<dim3(HWP / 256, 1,  NB), 256, 0, stream>>>(coef3, c4_w, c4_b, coef4, 3,   0);

  // 8) output = x + coef * y
  final_kernel<<<dim3(HWP / 256, CCH, NB), 256, 0, stream>>>(x, yc, coef4, outp);
}
